// CrossAttentionFusion_80427557585204
// MI455X (gfx1250) — matrix operand order repaired and checked by reading, not compiled, not run on hardware
//
#include <hip/hip_runtime.h>
#include <hip/hip_bf16.h>

// Problem constants (fixed by reference)
#define BB 16
#define CC 256
#define NN 2304   // 48*48

typedef __bf16 bf16x8  __attribute__((ext_vector_type(8)));
typedef __bf16 bf16x16 __attribute__((ext_vector_type(16)));
typedef float  v8f     __attribute__((ext_vector_type(8)));
typedef float  f32x8   __attribute__((ext_vector_type(8)));
typedef unsigned int u32x4 __attribute__((ext_vector_type(4)));
typedef int  i32x4 __attribute__((ext_vector_type(4)));
typedef int  i32x8 __attribute__((ext_vector_type(8)));

union AFrag { bf16x16 f; bf16x8 h[2]; };

__device__ __forceinline__ v8f vzero() {
  v8f x;
#pragma unroll
  for (int i = 0; i < 8; i++) x[i] = 0.f;
  return x;
}

__device__ __forceinline__ v8f wmma_bf16(bf16x16 a, bf16x16 b, v8f c) {
  // D = A(16x32) * B(32x16) + C, f32 accumulate
  return __builtin_amdgcn_wmma_f32_16x16x32_bf16(false, a, false, b, (short)0, c,
                                                 false, false);
}

// --- CDNA5 async data movement helpers -------------------------------------
// Per-lane 16B global -> LDS async copy (ASYNCcnt). LDS byte address is the
// low 32 bits of the generic pointer (LDS aperture maps addr[31:0]).
__device__ __forceinline__ void async_copy_b128(const void* g, void* l) {
  unsigned ldsoff = (unsigned)(size_t)l;
  asm volatile("global_load_async_to_lds_b128 %0, %1, off"
               :: "v"(ldsoff), "v"(g) : "memory");
}
__device__ __forceinline__ void wait_async0() {
  asm volatile("s_wait_asynccnt 0" ::: "memory");
}

#if defined(__AMDGCN__) && __has_builtin(__builtin_amdgcn_tensor_load_to_lds)
#define HAVE_TDM 1
// Issue a TDM load of one contiguous run of `n8` 8-byte elements from `gaddr`
// to LDS offset `ldsoff` (D# per CDNA5 ISA §8; 2D tensor with a single row).
// This toolchain uses the 6-argument builtin form:
//   (u32x4 g0, i32x8 g1, i32x4 g2, i32x4 g3, i32x8 g4, i32 cpol)
__device__ __forceinline__ void tdm_load_contig(unsigned long long gaddr,
                                                unsigned ldsoff, unsigned n8) {
  u32x4 g0;
  g0.x = 1u;                                               // count=1, user mode
  g0.y = ldsoff;                                           // lds_addr
  g0.z = (unsigned)(gaddr & 0xffffffffull);                // global_addr[31:0]
  g0.w = (unsigned)((gaddr >> 32) & 0x01ffffffull) | (2u << 30);  // [56:32]|type=2
  i32x8 g1;
  g1[0] = (int)(3u << 16);          // workgroup_mask=0, data_size=3 (8B)
  g1[1] = (int)((n8 & 0xffffu) << 16);   // tensor_dim0[15:0]
  g1[2] = (int)((n8 >> 16) | (1u << 16)); // tensor_dim0[31:16], tensor_dim1=1
  g1[3] = (int)((n8 & 0xffffu) << 16);   // tile_dim0
  g1[4] = 1;                        // tile_dim1=1
  g1[5] = (int)n8;                  // tensor_dim0_stride[31:0]
  g1[6] = (int)((n8 & 0xffffu) << 16);   // tensor_dim1_stride[15:0]
  g1[7] = 0;
  i32x4 g2 = {0, 0, 0, 0};
  i32x4 g3 = {0, 0, 0, 0};
  i32x8 g4 = {0, 0, 0, 0, 0, 0, 0, 0};
  __builtin_amdgcn_tensor_load_to_lds(g0, g1, g2, g3, g4, 0);
}
#endif

// ---------------------------------------------------------------------------
// Kernel 1: transpose [B,C,N] f32 -> [B,N,C] bf16  (LDS tiled, coalesced both ways)
// grid (N/32, C/32, 2*B), block (32,8)
// ---------------------------------------------------------------------------
__global__ void prep_kernel(const float* __restrict__ img, const float* __restrict__ msk,
                            __bf16* __restrict__ ImgT, __bf16* __restrict__ MskT) {
  __shared__ float tile[32][33];
  const int which = blockIdx.z >> 4;  // 0 = image, 1 = mask
  const int b     = blockIdx.z & 15;
  const float* src = which ? msk : img;
  __bf16*      dst = which ? MskT : ImgT;
  const int n0 = blockIdx.x * 32;
  const int c0 = blockIdx.y * 32;
  const int tx = threadIdx.x, ty = threadIdx.y;
#pragma unroll
  for (int i = 0; i < 32; i += 8) {
    const int c = c0 + ty + i;
    const int n = n0 + tx;
    tile[ty + i][tx] = src[((size_t)b * CC + c) * NN + n];
  }
  __syncthreads();
#pragma unroll
  for (int i = 0; i < 32; i += 8) {
    const int n = n0 + ty + i;
    const int c = c0 + tx;
    dst[((size_t)b * NN + n) * CC + c] = (__bf16)tile[tx][ty + i];
  }
}

// ---------------------------------------------------------------------------
// Kernel 2: convert weights f32 -> bf16 (row-major [o][c], K-innermost already)
// grid (256, 4), block 256
// ---------------------------------------------------------------------------
__global__ void convw_kernel(const float* __restrict__ Wq, const float* __restrict__ Wk,
                             const float* __restrict__ Wv, const float* __restrict__ Wo,
                             __bf16* __restrict__ oq, __bf16* __restrict__ ok,
                             __bf16* __restrict__ ov, __bf16* __restrict__ oo) {
  const int i = blockIdx.x * blockDim.x + threadIdx.x;  // < 65536
  const float* s;
  __bf16* d;
  switch (blockIdx.y) {
    case 0: s = Wq; d = oq; break;
    case 1: s = Wk; d = ok; break;
    case 2: s = Wv; d = ov; break;
    default: s = Wo; d = oo; break;
  }
  d[i] = (__bf16)s[i];
}

// ---------------------------------------------------------------------------
// Kernel 3: fused projections.
//   z=0: Qt[b,n,o] = (MskT[b,n,:]·Wq[o,:] + bq[o]) * C^-1/2   ([B,N,C] bf16)
//   z=1: Kt[b,n,o] =  ImgT[b,n,:]·Wk[o,:] + bk[o]             ([B,N,C] bf16)
//   z=2: V [b,o,m] =  ImgT[b,m,:]·Wv[o,:] + bv[o]             ([B,C,N] bf16)
// grid (B, N/64, 3), block 128 (4 waves, 16 rows each, full 256 outputs)
// ---------------------------------------------------------------------------
__global__ __launch_bounds__(128)
void proj_kernel(const __bf16* __restrict__ MskT, const __bf16* __restrict__ ImgT,
                 const __bf16* __restrict__ WqB, const __bf16* __restrict__ WkB,
                 const __bf16* __restrict__ WvB,
                 const float* __restrict__ bq, const float* __restrict__ bk,
                 const float* __restrict__ bv,
                 __bf16* __restrict__ Qt, __bf16* __restrict__ Kt,
                 __bf16* __restrict__ Vm) {
  const int b    = blockIdx.x;
  const int tgt  = blockIdx.z;  // 0=q, 1=k, 2=v
  const int wave = threadIdx.x >> 5;
  const int lane = threadIdx.x & 31;
  const int lo   = lane & 15;
  const bool hi  = lane >= 16;
  const int n0   = blockIdx.y * 64 + wave * 16;

  const __bf16* X    = (tgt == 0) ? MskT : ImgT;
  const __bf16* W    = (tgt == 0) ? WqB : (tgt == 1) ? WkB : WvB;
  const float*  bias = (tgt == 0) ? bq : (tgt == 1) ? bk : bv;

  const __bf16* xrow = X + ((size_t)b * NN + (n0 + lo)) * CC + (hi ? 16 : 0);
  const __bf16* wrow = W + (size_t)lo * CC + (hi ? 16 : 0);

  v8f acc[16];
#pragma unroll
  for (int j = 0; j < 16; j++) acc[j] = vzero();

#pragma unroll
  for (int k = 0; k < 8; k++) {
    AFrag a;
    a.h[0] = *(const bf16x8*)(xrow + 32 * k);
    a.h[1] = *(const bf16x8*)(xrow + 32 * k + 8);
    // double-buffer the B fragment so loads overlap WMMA issue
    bf16x16 cur = *(const bf16x16*)(wrow + 32 * k);
#pragma unroll
    for (int j = 0; j < 16; j++) {
      bf16x16 nxt;
      if (j < 15) nxt = *(const bf16x16*)(wrow + (size_t)(j + 1) * 16 * CC + 32 * k);
      acc[j] = wmma_bf16(a.f, cur, acc[j]);
      cur = nxt;
    }
  }

  const float scale = (tgt == 0) ? 0.0625f : 1.0f;  // C^-1/2 folded into Q
#pragma unroll
  for (int j = 0; j < 16; j++) {
    const int o = j * 16 + lo;
    const float bb = bias[o];
    v8f vals;
#pragma unroll
    for (int r = 0; r < 8; r++) vals[r] = (acc[j][r] + bb) * scale;
    if (tgt == 2) {
      // V [B,C,N]: lane's column o is fixed; rows r are consecutive m -> pack store
      bf16x8 pk;
#pragma unroll
      for (int r = 0; r < 8; r++) pk[r] = (__bf16)vals[r];
      *(bf16x8*)(Vm + ((size_t)b * CC + o) * NN + n0 + (hi ? 8 : 0)) = pk;
    } else {
      __bf16* dst = ((tgt == 0) ? Qt : Kt) + (size_t)b * NN * CC;
#pragma unroll
      for (int r = 0; r < 8; r++) {
        const int n = n0 + (hi ? 8 + r : r);
        dst[(size_t)n * CC + (j * 16 + lo)] = (__bf16)vals[r];
      }
    }
  }
}

// ---------------------------------------------------------------------------
// Kernel 4: flash attention.  Each wave owns 16 query rows and the full
// O[16 x 256] f32 accumulator; loops m in tiles of 32 with online softmax.
// K tile (contiguous 16KB) is fetched by the Tensor Data Mover (TENSORcnt);
// V tile rows use global_load_async_to_lds_b128 (ASYNCcnt).  P goes through
// padded LDS to become the next WMMA A-operand.
// grid (B, N/64), block 128 (4 waves)
// ---------------------------------------------------------------------------
__global__ __launch_bounds__(128)
void attn_kernel(const __bf16* __restrict__ Qt, const __bf16* __restrict__ Kt,
                 const __bf16* __restrict__ Vm, __bf16* __restrict__ Ot) {
  __shared__ __align__(32) __bf16 ldsK[32 * 256];   // [m=32][c=256]
  __shared__ __align__(32) __bf16 ldsV[256 * 32];   // [c=256][m=32]
  __shared__ __align__(16) __bf16 ldsP[4][16 * 40]; // per-wave P, padded stride 40

  const int b    = blockIdx.x;
  const int tid  = threadIdx.x;
  const int wave = tid >> 5;
  const int lane = tid & 31;
  const int lo   = lane & 15;
  const bool hi  = lane >= 16;
  const int n0   = blockIdx.y * 64 + wave * 16;

  // Q fragments for this wave's 16 rows, full K=256 (8 frags), kept in VGPRs.
  AFrag qf[8];
  const __bf16* qrow = Qt + ((size_t)b * NN + (n0 + lo)) * CC + (hi ? 16 : 0);
#pragma unroll
  for (int k = 0; k < 8; k++) {
    qf[k].h[0] = *(const bf16x8*)(qrow + 32 * k);
    qf[k].h[1] = *(const bf16x8*)(qrow + 32 * k + 8);
  }

  v8f acc[16];
#pragma unroll
  for (int j = 0; j < 16; j++) acc[j] = vzero();
  v8f mx, l;
#pragma unroll
  for (int r = 0; r < 8; r++) { mx[r] = -1e30f; l[r] = 0.f; }

  const __bf16* Kb = Kt + (size_t)b * NN * CC;
  const __bf16* Vb = Vm + (size_t)b * CC * NN;
  const float L2E = 1.44269504088896340736f;

  for (int m0 = 0; m0 < NN; m0 += 32) {
    __syncthreads();  // previous-iteration LDS reads complete

    // --- K tile: one contiguous 16KB run (rows m0..m0+31 of [N,C]) ----------
    const __bf16* ks = Kb + (size_t)m0 * CC;
#if defined(HAVE_TDM)
    if (wave == 0) {
      tdm_load_contig((unsigned long long)(size_t)ks, (unsigned)(size_t)ldsK,
                      (32u * 256u * 2u) / 8u /* = 2048 x 8B */);
    }
#else
#pragma unroll
    for (int i = 0; i < 8; i++) {
      const int chunk = tid + i * 128;  // 1024 x 16B
      async_copy_b128(ks + chunk * 8, ldsK + chunk * 8);
    }
#endif
    // --- V tile: 256 rows x 64B, global row stride N, async to LDS ----------
#pragma unroll
    for (int i = 0; i < 8; i++) {
      const int chunk = tid + i * 128;
      const int row = chunk >> 2;
      const int off = (chunk & 3) * 8;
      async_copy_b128(Vb + (size_t)row * NN + m0 + off, ldsV + row * 32 + off);
    }
    // prefetch next K tile while DMA fills this one
    if (m0 + 32 < NN)
      __builtin_prefetch(Kb + (size_t)(m0 + 32) * CC + tid * 64, 0, 1);

    wait_async0();
#if defined(HAVE_TDM)
    if (wave == 0) __builtin_amdgcn_s_wait_tensorcnt(0);
#endif
    __syncthreads();

    // --- S = Q * K^T for two 16x16 m-tiles ---
    v8f s0 = vzero(), s1 = vzero();
#pragma unroll
    for (int k = 0; k < 8; k++) {
      const bf16x16 b0 =
          *(const bf16x16*)(ldsK + (size_t)lo * 256 + 32 * k + (hi ? 16 : 0));
      const bf16x16 b1 =
          *(const bf16x16*)(ldsK + (size_t)(16 + lo) * 256 + 32 * k + (hi ? 16 : 0));
      s0 = wmma_bf16(qf[k].f, b0, s0);
      s1 = wmma_bf16(qf[k].f, b1, s1);
    }

    // --- online softmax (row stats via 16-lane shfl_xor butterflies) ---
    v8f mnew, alpha, p0, p1;
#pragma unroll
    for (int r = 0; r < 8; r++) {
      float v = fmaxf(s0[r], s1[r]);
#pragma unroll
      for (int off = 1; off < 16; off <<= 1) v = fmaxf(v, __shfl_xor(v, off));
      mnew[r] = fmaxf(mx[r], v);
      alpha[r] = exp2f((mx[r] - mnew[r]) * L2E);
    }
    mx = mnew;
#pragma unroll
    for (int r = 0; r < 8; r++) {
      p0[r] = exp2f((s0[r] - mx[r]) * L2E);
      p1[r] = exp2f((s1[r] - mx[r]) * L2E);
      float s = p0[r] + p1[r];
#pragma unroll
      for (int off = 1; off < 16; off <<= 1) s += __shfl_xor(s, off);
      l[r] = l[r] * alpha[r] + s;
    }
#pragma unroll
    for (int j = 0; j < 16; j++) acc[j] = acc[j] * alpha;

    // --- P -> LDS (transpose to A-operand layout) ---
    __bf16* Pw = ldsP[wave];
#pragma unroll
    for (int r = 0; r < 8; r++) {
      const int row = hi ? 8 + r : r;
      Pw[row * 40 + lo]      = (__bf16)p0[r];
      Pw[row * 40 + 16 + lo] = (__bf16)p1[r];
    }
    asm volatile("s_wait_dscnt 0" ::: "memory");  // same-wave DS RAW fence

    AFrag pa;
    const __bf16* pr = Pw + lo * 40 + (hi ? 16 : 0);
    pa.h[0] = *(const bf16x8*)(pr);
    pa.h[1] = *(const bf16x8*)(pr + 8);

    // --- O += P * V ---
#pragma unroll
    for (int j = 0; j < 16; j++) {
      const bf16x16 vb =
          *(const bf16x16*)(ldsV + (size_t)(j * 16 + lo) * 32 + (hi ? 16 : 0));
      acc[j] = wmma_bf16(pa.f, vb, acc[j]);
    }
  }

  // --- normalize and store Ot[b,n,c] bf16 ---
  v8f linv;
#pragma unroll
  for (int r = 0; r < 8; r++) linv[r] = 1.0f / l[r];
  __bf16* Ob = Ot + (size_t)b * NN * CC;
#pragma unroll
  for (int j = 0; j < 16; j++) {
    const int c = j * 16 + lo;
#pragma unroll
    for (int r = 0; r < 8; r++) {
      const int n = n0 + (hi ? 8 + r : r);
      Ob[(size_t)n * CC + c] = (__bf16)(acc[j][r] * linv[r]);
    }
  }
}

// ---------------------------------------------------------------------------
// Kernel 5: out[b,o,n] = resid[b,o,n] + bo[o] + sum_c Ot[b,n,c]*Wo[o,c]
// grid (B, N/64, C/64), block 128
// ---------------------------------------------------------------------------
__global__ __launch_bounds__(128)
void out_kernel(const __bf16* __restrict__ Ot, const __bf16* __restrict__ WoB,
                const float* __restrict__ bo, const float* __restrict__ resid,
                float* __restrict__ out) {
  const int b    = blockIdx.x;
  const int wave = threadIdx.x >> 5;
  const int lane = threadIdx.x & 31;
  const int lo   = lane & 15;
  const bool hi  = lane >= 16;
  const int n0   = blockIdx.y * 64 + wave * 16;
  const int o0   = blockIdx.z * 64;

  v8f acc[4];
#pragma unroll
  for (int j = 0; j < 4; j++) acc[j] = vzero();

  const __bf16* arow = Ot + ((size_t)b * NN + (n0 + lo)) * CC + (hi ? 16 : 0);
  const __bf16* wrow = WoB + (size_t)(o0 + lo) * CC + (hi ? 16 : 0);
#pragma unroll
  for (int k = 0; k < 8; k++) {
    AFrag a;
    a.h[0] = *(const bf16x8*)(arow + 32 * k);
    a.h[1] = *(const bf16x8*)(arow + 32 * k + 8);
    bf16x16 cur = *(const bf16x16*)(wrow + 32 * k);
#pragma unroll
    for (int j = 0; j < 4; j++) {
      bf16x16 nxt;
      if (j < 3) nxt = *(const bf16x16*)(wrow + (size_t)(j + 1) * 16 * CC + 32 * k);
      acc[j] = wmma_bf16(a.f, cur, acc[j]);
      cur = nxt;
    }
  }

#pragma unroll
  for (int j = 0; j < 4; j++) {
    const int o = o0 + j * 16 + lo;
    const float bb = bo[o];
    const size_t base = ((size_t)b * CC + o) * NN + n0 + (hi ? 8 : 0);
    const f32x8 rv = *(const f32x8*)(resid + base);
    f32x8 vals;
#pragma unroll
    for (int r = 0; r < 8; r++) vals[r] = acc[j][r] + bb + rv[r];
    *(f32x8*)(out + base) = vals;
  }
}

// ---------------------------------------------------------------------------
extern "C" void kernel_launch(void* const* d_in, const int* in_sizes, int n_in,
                              void* d_out, int out_size, void* d_ws, size_t ws_size,
                              hipStream_t stream) {
  (void)in_sizes; (void)n_in; (void)out_size; (void)ws_size;
  const float* img = (const float*)d_in[0];
  const float* msk = (const float*)d_in[1];
  const float* Wq  = (const float*)d_in[2];
  const float* bq  = (const float*)d_in[3];
  const float* Wk  = (const float*)d_in[4];
  const float* bk  = (const float*)d_in[5];
  const float* Wv  = (const float*)d_in[6];
  const float* bv  = (const float*)d_in[7];
  const float* Wo  = (const float*)d_in[8];
  const float* bo  = (const float*)d_in[9];

  char* ws = (char*)d_ws;
  const size_t S1 = (size_t)BB * NN * CC * sizeof(__bf16);  // one [B,N,C] bf16 buf
  __bf16* ImgT = (__bf16*)(ws + 0 * S1);
  __bf16* MskT = (__bf16*)(ws + 1 * S1);
  __bf16* Qt   = (__bf16*)(ws + 2 * S1);
  __bf16* Kt   = (__bf16*)(ws + 3 * S1);
  __bf16* Vm   = (__bf16*)(ws + 4 * S1);
  __bf16* WqB  = (__bf16*)(ws + 5 * S1);
  __bf16* WkB  = WqB + CC * CC;
  __bf16* WvB  = WkB + CC * CC;
  __bf16* WoB  = WvB + CC * CC;
  __bf16* Ot   = ImgT;  // safe alias: ImgT dead after proj_kernel

  prep_kernel<<<dim3(NN / 32, CC / 32, 2 * BB), dim3(32, 8), 0, stream>>>(
      img, msk, ImgT, MskT);
  convw_kernel<<<dim3(256, 4), 256, 0, stream>>>(Wq, Wk, Wv, Wo, WqB, WkB, WvB, WoB);
  proj_kernel<<<dim3(BB, NN / 64, 3), 128, 0, stream>>>(
      MskT, ImgT, WqB, WkB, WvB, bq, bk, bv, Qt, Kt, Vm);
  attn_kernel<<<dim3(BB, NN / 64), 128, 0, stream>>>(Qt, Kt, Vm, Ot);
  out_kernel<<<dim3(BB, NN / 64, CC / 64), 128, 0, stream>>>(Ot, WoB, bo, img,
                                                             (float*)d_out);
}
